// TopoformerMambaBlock_24404004176491
// MI455X (gfx1250) — matrix operand order repaired and checked by reading, not compiled, not run on hardware
//
#include <hip/hip_runtime.h>

#define DMODEL 1024
#define DSTATE 16
#define DCONV  4
#define DINNER 2048
#define DTRANK 64
#define BBATCH 2
#define LSEQ   2048
#define BL     (BBATCH * LSEQ)        // 4096 rows
#define NXZ    (2 * DINNER)           // 4096
#define NXDBL  (DTRANK + 2 * DSTATE)  // 96 (logical)
#define NXDBLP 128                    // padded to a multiple of 64

typedef __attribute__((ext_vector_type(16))) __bf16 bf16x16;
typedef __attribute__((ext_vector_type(8)))  __bf16 bf16x8;
typedef __attribute__((ext_vector_type(8)))  float  f32x8;

static __device__ __forceinline__ unsigned short f2bf(float f) {
  unsigned u = __float_as_uint(f);
  u += 0x7FFFu + ((u >> 16) & 1u);      // round-to-nearest-even
  return (unsigned short)(u >> 16);
}

// ---------------------------------------------------------------------------
// Weight convert+transpose: W [K,N] f32 -> Wt [N,K] bf16
// ---------------------------------------------------------------------------
__global__ void wtrans_kernel(const float* __restrict__ W,
                              unsigned short* __restrict__ Wt,
                              int K, int N) {
  size_t i = (size_t)blockIdx.x * blockDim.x + threadIdx.x;
  if (i >= (size_t)K * N) return;
  int n = (int)(i / K);
  int k = (int)(i - (size_t)n * K);
  Wt[i] = f2bf(W[(size_t)k * N + n]);
}

// ---------------------------------------------------------------------------
// LayerNorm: one wave32 per row of [BL, DMODEL]; writes f32 + bf16
// ---------------------------------------------------------------------------
__global__ void ln_kernel(const float* __restrict__ x,
                          const float* __restrict__ g,
                          const float* __restrict__ b,
                          float* __restrict__ hn,
                          unsigned short* __restrict__ hnb) {
  int wave = (int)((blockIdx.x * blockDim.x + threadIdx.x) >> 5);
  int lane = (int)(threadIdx.x & 31);
  if (wave >= BL) return;
  const float* row = x + (size_t)wave * DMODEL;
  float s = 0.f, s2 = 0.f;
  for (int i = lane; i < DMODEL; i += 32) {
    float v = row[i];
    s += v; s2 += v * v;
  }
  for (int off = 16; off > 0; off >>= 1) {
    s  += __shfl_xor(s,  off, 32);
    s2 += __shfl_xor(s2, off, 32);
  }
  float mu  = s * (1.f / DMODEL);
  float var = s2 * (1.f / DMODEL) - mu * mu;
  float inv = rsqrtf(var + 1e-5f);
  for (int i = lane; i < DMODEL; i += 32) {
    float v = (row[i] - mu) * inv * g[i] + b[i];
    hn [(size_t)wave * DMODEL + i] = v;
    hnb[(size_t)wave * DMODEL + i] = f2bf(v);
  }
}

// ---------------------------------------------------------------------------
// WMMA GEMM, 2x4 register blocking: each wave32 computes a 32x64 output
// region = 8 accumulator tiles; 8 v_wmma per K-step against 12 b128 loads.
// D = A[M,K](bf16) x Bt[N,K](bf16)^T + fused epilogue.
// Requires: M % 32 == 0, N % 64 == 0, K % 32 == 0 (host pads N).
// Tile indices are readfirstlane'd so all tile control flow is scalar and
// EXEC is never modified around WMMA.
// ---------------------------------------------------------------------------
__global__ __launch_bounds__(256) void
wmma_gemm_kernel(const __bf16* __restrict__ A,
                 const __bf16* __restrict__ Bt,
                 const float* __restrict__ bias,       // [N] or null
                 const float* __restrict__ add_src,    // [M,N] or null
                 const float* __restrict__ mul_src,    // [M,N] or null
                 const float* __restrict__ res_src,    // [M,N] or null
                 const float* __restrict__ scale_ptr,  // scalar or null
                 float* __restrict__ out_f32,          // [M,N] or null
                 unsigned short* __restrict__ out_bf16,// [M,N] or null
                 int M, int N, int K, int do_softplus) {
  const int wave = __builtin_amdgcn_readfirstlane(
      (int)((blockIdx.x * blockDim.x + threadIdx.x) >> 5));
  const int lane = (int)(threadIdx.x & 31);
  const int ntile4 = N >> 6;                    // 64-wide column groups
  const int tm = wave / ntile4;
  const int tn = wave - tm * ntile4;
  if (tm * 32 >= M) return;                     // scalar branch

  const int half = lane >> 4;                   // K-half selector (contiguous k = 16*half + s)
  const int sub  = lane & 15;                   // A: M row / B: N col

  // A fragment rows for the two M tiles
  const __bf16* __restrict__ pa0 = A + (size_t)(tm * 32 + sub) * K + half * 16;
  const __bf16* __restrict__ pa1 = pa0 + (size_t)16 * K;
  // B fragment cols for the four N tiles
  const __bf16* __restrict__ pb0 = Bt + (size_t)(tn * 64 + sub) * K + half * 16;
  const __bf16* __restrict__ pb1 = pb0 + (size_t)16 * K;
  const __bf16* __restrict__ pb2 = pb0 + (size_t)32 * K;
  const __bf16* __restrict__ pb3 = pb0 + (size_t)48 * K;

  f32x8 acc[2][4];
#pragma unroll
  for (int mi = 0; mi < 2; ++mi)
#pragma unroll
    for (int j = 0; j < 4; ++j)
#pragma unroll
      for (int r = 0; r < 8; ++r) acc[mi][j][r] = 0.f;

  for (int kb = 0; kb < K; kb += 32) {
    // A fragments: VGPR0-7 <- K=half*16+0..15 (matches B)
    bf16x8 a00 = *(const bf16x8*)(pa0 + kb);
    bf16x8 a01 = *(const bf16x8*)(pa0 + kb + 8);
    bf16x8 a10 = *(const bf16x8*)(pa1 + kb);
    bf16x8 a11 = *(const bf16x8*)(pa1 + kb + 8);
    // B fragments: VGPR0-7 <- K=half*16+0..15 (contiguous in [N,K])
    bf16x16 bm0 = *(const bf16x16*)(pb0 + kb);
    bf16x16 bm1 = *(const bf16x16*)(pb1 + kb);
    bf16x16 bm2 = *(const bf16x16*)(pb2 + kb);
    bf16x16 bm3 = *(const bf16x16*)(pb3 + kb);

    // Unguarded speculative near-cache prefetch of next K slab (fault-free).
    __builtin_prefetch(pa0 + kb + 32, 0, 3);
    __builtin_prefetch(pa1 + kb + 32, 0, 3);
    __builtin_prefetch(pb0 + kb + 32, 0, 3);
    __builtin_prefetch(pb1 + kb + 32, 0, 3);
    __builtin_prefetch(pb2 + kb + 32, 0, 3);
    __builtin_prefetch(pb3 + kb + 32, 0, 3);

    bf16x16 am0, am1;
#pragma unroll
    for (int i = 0; i < 8; ++i) {
      am0[i] = a00[i]; am0[8 + i] = a01[i];
      am1[i] = a10[i]; am1[8 + i] = a11[i];
    }

    acc[0][0] = __builtin_amdgcn_wmma_f32_16x16x32_bf16(false, am0, false, bm0, (short)0, acc[0][0], false, false);
    acc[1][0] = __builtin_amdgcn_wmma_f32_16x16x32_bf16(false, am1, false, bm0, (short)0, acc[1][0], false, false);
    acc[0][1] = __builtin_amdgcn_wmma_f32_16x16x32_bf16(false, am0, false, bm1, (short)0, acc[0][1], false, false);
    acc[1][1] = __builtin_amdgcn_wmma_f32_16x16x32_bf16(false, am1, false, bm1, (short)0, acc[1][1], false, false);
    acc[0][2] = __builtin_amdgcn_wmma_f32_16x16x32_bf16(false, am0, false, bm2, (short)0, acc[0][2], false, false);
    acc[1][2] = __builtin_amdgcn_wmma_f32_16x16x32_bf16(false, am1, false, bm2, (short)0, acc[1][2], false, false);
    acc[0][3] = __builtin_amdgcn_wmma_f32_16x16x32_bf16(false, am0, false, bm3, (short)0, acc[0][3], false, false);
    acc[1][3] = __builtin_amdgcn_wmma_f32_16x16x32_bf16(false, am1, false, bm3, (short)0, acc[1][3], false, false);
  }

  const float scl = scale_ptr ? scale_ptr[0] : 1.0f;
#pragma unroll
  for (int j = 0; j < 4; ++j) {
    const int   gn = tn * 64 + j * 16 + sub;
    const float bv = bias ? bias[gn] : 0.0f;
#pragma unroll
    for (int mi = 0; mi < 2; ++mi) {
#pragma unroll
      for (int r = 0; r < 8; ++r) {
        // D layout: VGPR r -> M = r (+8 for lanes 16-31)
        int gm = tm * 32 + mi * 16 + r + half * 8;
        float v = acc[mi][j][r] + bv;
        if (do_softplus) v = (v > 20.f) ? v : log1pf(__expf(v));
        size_t o = (size_t)gm * N + gn;
        if (add_src) v += add_src[o];
        if (mul_src) v *= mul_src[o];
        if (res_src) v = res_src[o] + scl * v;
        if (out_f32)  out_f32[o]  = v;
        if (out_bf16) out_bf16[o] = f2bf(v);
      }
    }
  }
}

// ---------------------------------------------------------------------------
// Causal depthwise conv (d_conv=4) + bias + SiLU on u = xz[:, :DINNER]
// ---------------------------------------------------------------------------
__global__ void conv_silu_kernel(const float* __restrict__ xz,
                                 const float* __restrict__ cw,   // [DINNER,4]
                                 const float* __restrict__ cb,   // [DINNER]
                                 float* __restrict__ uc,
                                 unsigned short* __restrict__ ucb) {
  size_t i = (size_t)blockIdx.x * blockDim.x + threadIdx.x;
  if (i >= (size_t)BL * DINNER) return;
  int d   = (int)(i % DINNER);
  int row = (int)(i / DINNER);
  int t   = row % LSEQ;
  int bb  = row / LSEQ;
  float acc = cb[d];
#pragma unroll
  for (int j = 0; j < DCONV; ++j) {
    int tt = t - (DCONV - 1) + j;
    if (tt >= 0)
      acc += xz[(size_t)(bb * LSEQ + tt) * NXZ + d] * cw[d * DCONV + j];
  }
  float s = acc / (1.f + __expf(-acc));   // SiLU
  uc[i]  = s;
  ucb[i] = f2bf(s);
}

// dt slice of x_dbl [BL,128-padded] cols 0..63 -> bf16 [BL,64]
__global__ void dt_slice_kernel(const float* __restrict__ xdbl,
                                unsigned short* __restrict__ dtb) {
  int i = blockIdx.x * blockDim.x + threadIdx.x;
  if (i >= BL * DTRANK) return;
  int m = i / DTRANK, c = i - m * DTRANK;
  dtb[i] = f2bf(xdbl[(size_t)m * NXDBLP + c]);
}

// ---------------------------------------------------------------------------
// Selective scan: one lane per (b, d_inner); 16-state recurrence in registers.
// Fuses +u*D, *silu(z); emits bf16 A-operand for out_proj.
// ---------------------------------------------------------------------------
__global__ void scan_kernel(const float* __restrict__ delta,  // [BL,DINNER]
                            const float* __restrict__ uc,     // [BL,DINNER]
                            const float* __restrict__ xz,     // [BL,NXZ]; z at +DINNER
                            const float* __restrict__ xdbl,   // [BL,128]; B@64, C@80
                            const float* __restrict__ A_log,  // [DINNER,16]
                            const float* __restrict__ Dp,     // [DINNER]
                            unsigned short* __restrict__ yb)  // [BL,DINNER]
{
  int tid = blockIdx.x * blockDim.x + threadIdx.x;   // 0 .. B*DINNER-1
  if (tid >= BBATCH * DINNER) return;
  int b = tid / DINNER;
  int d = tid - b * DINNER;
  float Av[DSTATE], h[DSTATE];
#pragma unroll
  for (int n = 0; n < DSTATE; ++n) {
    Av[n] = -__expf(A_log[d * DSTATE + n]);
    h[n]  = 0.f;
  }
  float Dd = Dp[d];
  for (int t = 0; t < LSEQ; ++t) {
    size_t row = (size_t)(b * LSEQ + t);
    float dv = delta[row * DINNER + d];
    float uv = uc[row * DINNER + d];
    const float* bc = xdbl + row * NXDBLP;
    float dBu = dv * uv;
    float acc = 0.f;
#pragma unroll
    for (int n = 0; n < DSTATE; ++n) {
      float dA = __expf(dv * Av[n]);
      h[n] = dA * h[n] + dBu * bc[DTRANK + n];
      acc += h[n] * bc[DTRANK + DSTATE + n];
    }
    float y = acc + uv * Dd;
    float zv = xz[row * NXZ + DINNER + d];
    y *= zv / (1.f + __expf(-zv));         // * silu(z)
    yb[row * DINNER + d] = f2bf(y);
  }
}

// ---------------------------------------------------------------------------
// Host side
// ---------------------------------------------------------------------------
static inline void* ws_bump(char*& p, size_t bytes) {
  void* r = (void*)p;
  p += (bytes + 255) & ~(size_t)255;
  return r;
}

static inline void launch_gemm(const void* A, const void* Bt, const float* bias,
                               const float* add_src, const float* mul_src,
                               const float* res_src, const float* scale_ptr,
                               float* out_f32, unsigned short* out_bf16,
                               int M, int N, int K, int softplus, hipStream_t stream) {
  int waves   = (M / 32) * (N / 64);
  int threads = 256;
  int blocks  = (waves * 32 + threads - 1) / threads;
  wmma_gemm_kernel<<<blocks, threads, 0, stream>>>(
      (const __bf16*)A, (const __bf16*)Bt, bias, add_src, mul_src, res_src,
      scale_ptr, out_f32, out_bf16, M, N, K, softplus);
}

extern "C" void kernel_launch(void* const* d_in, const int* in_sizes, int n_in,
                              void* d_out, int out_size, void* d_ws, size_t ws_size,
                              hipStream_t stream) {
  (void)in_sizes; (void)n_in; (void)out_size; (void)ws_size;

  const float* x        = (const float*)d_in[0];
  const float* ln_g     = (const float*)d_in[1];
  const float* ln_b     = (const float*)d_in[2];
  const float* sq_w     = (const float*)d_in[3];
  const float* sq_b     = (const float*)d_in[4];
  const float* srw_w    = (const float*)d_in[5];
  const float* srw_b    = (const float*)d_in[6];
  const float* res_scale= (const float*)d_in[7];
  const float* in_proj_w= (const float*)d_in[8];
  const float* conv_w   = (const float*)d_in[9];
  const float* conv_b   = (const float*)d_in[10];
  const float* x_proj_w = (const float*)d_in[11];
  const float* dt_w     = (const float*)d_in[12];
  const float* dt_b     = (const float*)d_in[13];
  const float* A_log    = (const float*)d_in[14];
  const float* Dp       = (const float*)d_in[15];
  const float* out_proj_w = (const float*)d_in[16];

  char* p = (char*)d_ws;
  unsigned short* sqT  = (unsigned short*)ws_bump(p, (size_t)DMODEL * DMODEL * 2);
  unsigned short* inT  = (unsigned short*)ws_bump(p, (size_t)NXZ    * DMODEL * 2);
  unsigned short* xpT  = (unsigned short*)ws_bump(p, (size_t)NXDBLP * DINNER * 2); // padded
  unsigned short* dtT  = (unsigned short*)ws_bump(p, (size_t)DINNER * DTRANK * 2);
  unsigned short* opT  = (unsigned short*)ws_bump(p, (size_t)DMODEL * DINNER * 2);
  unsigned short* srT  = (unsigned short*)ws_bump(p, (size_t)DMODEL * DMODEL * 2);
  float*          hn   = (float*)         ws_bump(p, (size_t)BL * DMODEL * 4);
  unsigned short* hnb  = (unsigned short*)ws_bump(p, (size_t)BL * DMODEL * 2);
  unsigned short* h2b  = (unsigned short*)ws_bump(p, (size_t)BL * DMODEL * 2);
  float*          xz   = (float*)         ws_bump(p, (size_t)BL * NXZ * 4);
  float*          uc   = (float*)         ws_bump(p, (size_t)BL * DINNER * 4);
  unsigned short* ucb  = (unsigned short*)ws_bump(p, (size_t)BL * DINNER * 2);
  float*          xdbl = (float*)         ws_bump(p, (size_t)BL * NXDBLP * 4);    // padded
  unsigned short* dtb  = (unsigned short*)ws_bump(p, (size_t)BL * DTRANK * 2);
  float*          delta= (float*)         ws_bump(p, (size_t)BL * DINNER * 4);
  unsigned short* yb   = (unsigned short*)ws_bump(p, (size_t)BL * DINNER * 2);
  float*          ym   = (float*)         ws_bump(p, (size_t)BL * DMODEL * 4);
  unsigned short* ymb  = (unsigned short*)ws_bump(p, (size_t)BL * DMODEL * 2);

  const int T = 256;
  auto blk = [](size_t n, int t) { return (int)((n + t - 1) / t); };

  // --- weight transpose+convert to bf16 [N,K]; x_proj padded to 128 rows ---
  hipMemsetAsync(xpT, 0, (size_t)NXDBLP * DINNER * 2, stream);
  wtrans_kernel<<<blk((size_t)DMODEL * DMODEL, T), T, 0, stream>>>(sq_w,     sqT, DMODEL, DMODEL);
  wtrans_kernel<<<blk((size_t)DMODEL * NXZ,    T), T, 0, stream>>>(in_proj_w,inT, DMODEL, NXZ);
  wtrans_kernel<<<blk((size_t)DINNER * NXDBL,  T), T, 0, stream>>>(x_proj_w, xpT, DINNER, NXDBL);
  wtrans_kernel<<<blk((size_t)DTRANK * DINNER, T), T, 0, stream>>>(dt_w,     dtT, DTRANK, DINNER);
  wtrans_kernel<<<blk((size_t)DINNER * DMODEL, T), T, 0, stream>>>(out_proj_w,opT, DINNER, DMODEL);
  wtrans_kernel<<<blk((size_t)DMODEL * DMODEL, T), T, 0, stream>>>(srw_w,    srT, DMODEL, DMODEL);

  // --- hn = layernorm(x) ---
  ln_kernel<<<blk((size_t)BL * 32, T), T, 0, stream>>>(x, ln_g, ln_b, hn, hnb);

  // --- h2 = hn + hn @ sq_w + sq_b  (bf16 out only) ---
  launch_gemm(hnb, sqT, sq_b, hn, nullptr, nullptr, nullptr,
              nullptr, h2b, BL, DMODEL, DMODEL, 0, stream);

  // --- xz = h2 @ in_proj_w ---
  launch_gemm(h2b, inT, nullptr, nullptr, nullptr, nullptr, nullptr,
              xz, nullptr, BL, NXZ, DMODEL, 0, stream);

  // --- u = silu(causal_dwconv(xz[:, :DINNER]) + conv_b) ---
  conv_silu_kernel<<<blk((size_t)BL * DINNER, T), T, 0, stream>>>(xz, conv_w, conv_b, uc, ucb);

  // --- x_dbl = u @ x_proj_w  [BL,128-padded] ---
  launch_gemm(ucb, xpT, nullptr, nullptr, nullptr, nullptr, nullptr,
              xdbl, nullptr, BL, NXDBLP, DINNER, 0, stream);
  dt_slice_kernel<<<blk((size_t)BL * DTRANK, T), T, 0, stream>>>(xdbl, dtb);

  // --- delta = softplus(dt @ dt_w + dt_b) ---
  launch_gemm(dtb, dtT, dt_b, nullptr, nullptr, nullptr, nullptr,
              delta, nullptr, BL, DINNER, DTRANK, 1, stream);

  // --- selective scan + *silu(z) + u*D -> yb (bf16) ---
  scan_kernel<<<blk((size_t)BBATCH * DINNER, T), T, 0, stream>>>(
      delta, uc, xz, xdbl, A_log, Dp, yb);

  // --- ym = y @ out_proj_w ---
  launch_gemm(yb, opT, nullptr, nullptr, nullptr, nullptr, nullptr,
              ym, ymb, BL, DMODEL, DINNER, 0, stream);

  // --- out = x + res_scale * (ym * (ym @ srw_w + srw_b)) ---
  launch_gemm(ymb, srT, srw_b, nullptr, ym, x, res_scale,
              (float*)d_out, nullptr, BL, DMODEL, DMODEL, 0, stream);
}
